// TransformerLayer_22428319220156
// MI455X (gfx1250) — matrix operand order repaired and checked by reading, not compiled, not run on hardware
//
#include <hip/hip_runtime.h>

// ---------------------------------------------------------------------------
// Problem constants (match reference)
// ---------------------------------------------------------------------------
constexpr int kB = 4, kS = 1024, kD = 1024, kH = 16, kDH = 64, kF = 4096;
constexpr float kEps = 1e-3f;

// ---------------------------------------------------------------------------
// WMMA types (gfx1250, wave32): v_wmma_f32_16x16x32_bf16
// ---------------------------------------------------------------------------
typedef __attribute__((ext_vector_type(16))) __bf16 v16bf;
typedef __attribute__((ext_vector_type(8)))  float  v8f;

union FragBF {
    v16bf v;
    uint4 q[2];
};

// ---------------------------------------------------------------------------
// CDNA5 async global->LDS copy (ASYNCcnt path, no VGPR data movement).
// LDS operand is the wave-relative LDS byte offset (low 32 bits of the
// generic pointer to __shared__, per the flat->LDS aperture truncation).
// ---------------------------------------------------------------------------
__device__ __forceinline__ void async_ld_b128(const void* gptr, void* lptr) {
    unsigned lds = (unsigned)(uintptr_t)lptr;
    asm volatile("global_load_async_to_lds_b128 %0, %1, off"
                 :: "v"(lds), "v"(gptr)
                 : "memory");
}
__device__ __forceinline__ void wait_async0() {
    asm volatile("s_wait_asynccnt 0x0" ::: "memory");
}

// ---------------------------------------------------------------------------
// fp32 -> bf16 conversion (weights)
// ---------------------------------------------------------------------------
__global__ __launch_bounds__(256)
void cvt_bf16_kernel(const float* __restrict__ in, __bf16* __restrict__ out, int n) {
    int i = blockIdx.x * blockDim.x + threadIdx.x;
    int stride = gridDim.x * blockDim.x;
    for (; i < n; i += stride) out[i] = (__bf16)in[i];
}

// ---------------------------------------------------------------------------
// LayerNorm over D=1024, one row per 256-thread block, bf16 output
// ---------------------------------------------------------------------------
__global__ __launch_bounds__(256)
void ln_bf16_kernel(const float* __restrict__ x, const float* __restrict__ g,
                    const float* __restrict__ be, __bf16* __restrict__ out) {
    __shared__ float red[256];
    const int row = blockIdx.x;
    const int tid = threadIdx.x;
    const float* xr = x + (size_t)row * kD;

    float v[4];
    float s = 0.f;
#pragma unroll
    for (int i = 0; i < 4; ++i) { v[i] = xr[tid + i * 256]; s += v[i]; }
    red[tid] = s;
    __syncthreads();
    for (int st = 128; st > 0; st >>= 1) {
        if (tid < st) red[tid] += red[tid + st];
        __syncthreads();
    }
    float mu = red[0] * (1.0f / kD);
    __syncthreads();

    float vs = 0.f;
#pragma unroll
    for (int i = 0; i < 4; ++i) { float d = v[i] - mu; vs += d * d; }
    red[tid] = vs;
    __syncthreads();
    for (int st = 128; st > 0; st >>= 1) {
        if (tid < st) red[tid] += red[tid + st];
        __syncthreads();
    }
    float rstd = rsqrtf(red[0] * (1.0f / kD) + kEps);

#pragma unroll
    for (int i = 0; i < 4; ++i) {
        int c = tid + i * 256;
        out[(size_t)row * kD + c] = (__bf16)((v[i] - mu) * rstd * g[c] + be[c]);
    }
}

// ---------------------------------------------------------------------------
// Tiled bf16 WMMA GEMM: out = epilogue(A[M,K] @ B[K,N] + bias)
//   WG: 256 threads = 8 waves; tile 128(M) x 128(N); K-step 32.
//   Each wave: 32x64 = 2x4 accumulators of 16x16.
//   A tile staged via async global->LDS DMA; B tile via registers (transpose).
// ---------------------------------------------------------------------------
constexpr int EPI_BF16_SCALE     = 0;  // out bf16 = (acc+bias)*scale
constexpr int EPI_RESID_F32      = 1;  // out f32  = acc+bias+resid
constexpr int EPI_GELU_BF16      = 2;  // out bf16 = gelu_exact(acc+bias)
constexpr int EPI_RESID_MASK_F32 = 3;  // out f32  = (acc+bias+resid)*(1-pad[row])

constexpr int GBM = 128, GBN = 128, GBK = 32;
constexpr int GLD = GBK + 8;  // padded LDS stride (bf16 elems)

template <int EPI>
__global__ __launch_bounds__(256)
void gemm_bf16_kernel(const __bf16* __restrict__ A, const __bf16* __restrict__ Bw,
                      const float* __restrict__ bias,
                      const float* __restrict__ resid,
                      const float* __restrict__ rowmask,
                      void* __restrict__ outp,
                      int M, int N, int K, float scale) {
    __shared__ __bf16 As[GBM * GLD];
    __shared__ __bf16 BsT[GBN * GLD];  // transposed: BsT[n][k]

    const int tid  = threadIdx.x;
    const int lane = tid & 31;
    const int wave = tid >> 5;
    const int wm   = wave & 3;   // 0..3 -> M offset wm*32
    const int wn   = wave >> 2;  // 0..1 -> N offset wn*64
    const int lrow = lane & 15;
    const int kb8  = (lane < 16) ? 0 : 8;
    const int kb16 = (lane < 16) ? 0 : 16;
    const int hi8  = (lane >= 16) ? 8 : 0;

    const int m0 = blockIdx.y * GBM;
    const int n0 = blockIdx.x * GBN;

    v8f acc[2][4] = {};

    for (int k0 = 0; k0 < K; k0 += GBK) {
        // A tile: 128 x 32, async DMA straight into LDS
#pragma unroll
        for (int it = 0; it < 2; ++it) {
            int idx = tid + it * 256;        // 0..511
            int row = idx >> 2;
            int seg = idx & 3;
            async_ld_b128(A + (size_t)(m0 + row) * K + k0 + seg * 8,
                          &As[row * GLD + seg * 8]);
        }
        // B tile: 32 x 128, loaded 16B then scattered transposed into LDS
#pragma unroll
        for (int it = 0; it < 2; ++it) {
            int idx  = tid + it * 256;       // 0..511
            int krow = idx >> 4;             // 0..31
            int seg  = idx & 15;             // 0..15
            uint4 pack = *reinterpret_cast<const uint4*>(
                Bw + (size_t)(k0 + krow) * N + n0 + seg * 8);
            const __bf16* hv = reinterpret_cast<const __bf16*>(&pack);
#pragma unroll
            for (int j = 0; j < 8; ++j)
                BsT[(seg * 8 + j) * GLD + krow] = hv[j];
        }
        wait_async0();
        __syncthreads();

        FragBF a[2], b[4];
#pragma unroll
        for (int mi = 0; mi < 2; ++mi) {
            const __bf16* p = &As[(wm * 32 + mi * 16 + lrow) * GLD + kb8];
            a[mi].q[0] = *reinterpret_cast<const uint4*>(p);
            a[mi].q[1] = *reinterpret_cast<const uint4*>(p + 16);
        }
#pragma unroll
        for (int ni = 0; ni < 4; ++ni) {
            const __bf16* p = &BsT[(wn * 64 + ni * 16 + lrow) * GLD + kb8];
            b[ni].q[0] = *reinterpret_cast<const uint4*>(p);
            b[ni].q[1] = *reinterpret_cast<const uint4*>(p + 16);
        }
#pragma unroll
        for (int mi = 0; mi < 2; ++mi)
#pragma unroll
            for (int ni = 0; ni < 4; ++ni)
                acc[mi][ni] = __builtin_amdgcn_wmma_f32_16x16x32_bf16(
                    false, a[mi].v, false, b[ni].v, (short)0, acc[mi][ni], false, false);
        __syncthreads();
    }

    // Epilogue. C layout: lanes 0-15 -> N=lane, M=r; lanes 16-31 -> N=lane-16, M=r+8.
#pragma unroll
    for (int mi = 0; mi < 2; ++mi) {
#pragma unroll
        for (int ni = 0; ni < 4; ++ni) {
#pragma unroll
            for (int r = 0; r < 8; ++r) {
                int row = m0 + wm * 32 + mi * 16 + r + hi8;
                int col = n0 + wn * 64 + ni * 16 + lrow;
                float v = acc[mi][ni][r] + bias[col];
                size_t o = (size_t)row * N + col;
                if constexpr (EPI == EPI_BF16_SCALE) {
                    ((__bf16*)outp)[o] = (__bf16)(v * scale);
                } else if constexpr (EPI == EPI_RESID_F32) {
                    ((float*)outp)[o] = v + resid[o];
                } else if constexpr (EPI == EPI_GELU_BF16) {
                    float g = 0.5f * v * (1.0f + erff(v * 0.70710678118654752f));
                    ((__bf16*)outp)[o] = (__bf16)g;
                } else {  // EPI_RESID_MASK_F32
                    ((float*)outp)[o] = (v + resid[o]) * (1.0f - rowmask[row]);
                }
            }
        }
    }
}

// ---------------------------------------------------------------------------
// Attention kernel: one WG per (b, h, 64-query tile).
// Full 64x1024 fp32 logits block lives in LDS (~257KB of the 320KB WGP LDS),
// softmax is a pure-LDS wave32 pass, context GEMM streams V through LDS.
// Q/K tiles staged via async global->LDS DMA.
// ---------------------------------------------------------------------------
constexpr int LGS = 1028;  // logits LDS stride (floats), padded vs 64 banks
constexpr int TS  = 72;    // bf16 tile stride (64 + 8)
constexpr unsigned ATTN_SMEM =
    64u * LGS * sizeof(float) + 3u * 64u * TS * sizeof(__bf16) + 64u * sizeof(float);

__global__ __launch_bounds__(256)
void attn_kernel(const __bf16* __restrict__ Q, const __bf16* __restrict__ Kb,
                 const __bf16* __restrict__ Vb,
                 const float* __restrict__ abias, const float* __restrict__ amask,
                 const float* __restrict__ pad, __bf16* __restrict__ ctx) {
    extern __shared__ char smem[];
    float*  Lg = reinterpret_cast<float*>(smem);
    __bf16* Qs = reinterpret_cast<__bf16*>(smem + (size_t)64 * LGS * sizeof(float));
    __bf16* Ks = Qs + 64 * TS;
    __bf16* VT = Ks + 64 * TS;  // transposed V tile: VT[dh][key_local]
    float*  rowInv = reinterpret_cast<float*>(VT + 64 * TS);

    const int tid  = threadIdx.x;
    const int lane = tid & 31;
    const int wave = tid >> 5;
    const int lrow = lane & 15;
    const int kb8  = (lane < 16) ? 0 : 8;
    const int kb16 = (lane < 16) ? 0 : 16;
    const int hi8  = (lane >= 16) ? 8 : 0;

    const int bx = blockIdx.x;
    const int qt = bx & 15;          // S/64 = 16 query tiles
    const int h  = (bx >> 4) & 15;   // head
    const int b  = bx >> 8;          // batch
    const int q0 = qt * 64;

    // ---- stage Q tile (64 x 64 bf16) via async DMA ----
#pragma unroll
    for (int it = 0; it < 2; ++it) {
        int idx = tid + it * 256;
        int row = idx >> 3, seg = idx & 7;
        async_ld_b128(Q + (size_t)(b * kS + q0 + row) * kD + h * kDH + seg * 8,
                      &Qs[row * TS + seg * 8]);
    }

    // ---- phase 1: logits = Q*scale @ K^T + bias + mask ----
    for (int kt = 0; kt < 16; ++kt) {
#pragma unroll
        for (int it = 0; it < 2; ++it) {
            int idx = tid + it * 256;
            int row = idx >> 3, seg = idx & 7;
            async_ld_b128(Kb + (size_t)(b * kS + kt * 64 + row) * kD + h * kDH + seg * 8,
                          &Ks[row * TS + seg * 8]);
        }
        wait_async0();
        __syncthreads();

#pragma unroll
        for (int sti = 0; sti < 2; ++sti) {
            int st = wave + sti * 8;     // 16 subtiles of 16x16
            int qm = st >> 2, kn = st & 3;
            v8f c = {};
#pragma unroll
            for (int ks = 0; ks < kDH; ks += 32) {
                FragBF a, bb;
                const __bf16* pa = &Qs[(qm * 16 + lrow) * TS + ks + kb8];
                a.q[0] = *reinterpret_cast<const uint4*>(pa);
                a.q[1] = *reinterpret_cast<const uint4*>(pa + 16);
                const __bf16* pb = &Ks[(kn * 16 + lrow) * TS + ks + kb8];
                bb.q[0] = *reinterpret_cast<const uint4*>(pb);
                bb.q[1] = *reinterpret_cast<const uint4*>(pb + 16);
                c = __builtin_amdgcn_wmma_f32_16x16x32_bf16(
                    false, a.v, false, bb.v, (short)0, c, false, false);
            }
#pragma unroll
            for (int r = 0; r < 8; ++r) {
                int qrow = qm * 16 + r + hi8;
                int qg   = q0 + qrow;
                int kg   = kt * 64 + kn * 16 + lrow;
                float keep = (1.0f - pad[b * kS + qg]) * (1.0f - pad[b * kS + kg]);
                float m = amask[((size_t)(b * kS) + qg) * kS + kg] * keep;
                float v = c[r] + abias[(((size_t)(b * kH) + h) * kS + qg) * kS + kg] -
                          1e9f * (1.0f - m);
                Lg[qrow * LGS + kg] = v;
            }
        }
        __syncthreads();
    }

    // ---- phase 2: softmax over full key dim, in LDS ----
    for (int r8 = 0; r8 < 8; ++r8) {
        int row = wave * 8 + r8;
        float mx = -3.0e38f;
        for (int c0 = lane; c0 < kS; c0 += 32) mx = fmaxf(mx, Lg[row * LGS + c0]);
#pragma unroll
        for (int off = 16; off > 0; off >>= 1) mx = fmaxf(mx, __shfl_xor(mx, off, 32));
        float s = 0.f;
        for (int c0 = lane; c0 < kS; c0 += 32) {
            float e = __expf(Lg[row * LGS + c0] - mx);
            Lg[row * LGS + c0] = e;
            s += e;
        }
#pragma unroll
        for (int off = 16; off > 0; off >>= 1) s += __shfl_xor(s, off, 32);
        if (lane == 0) rowInv[row] = 1.0f / s;
    }
    __syncthreads();

    // ---- phase 3: ctx = P @ V (1/rowsum folded into epilogue) ----
    v8f o[2] = {};
    for (int kc = 0; kc < 16; ++kc) {
#pragma unroll
        for (int it = 0; it < 2; ++it) {
            int idx = tid + it * 256;
            int row = idx >> 3, seg = idx & 7;  // row = key local
            uint4 pack = *reinterpret_cast<const uint4*>(
                Vb + (size_t)(b * kS + kc * 64 + row) * kD + h * kDH + seg * 8);
            const __bf16* hv = reinterpret_cast<const __bf16*>(&pack);
#pragma unroll
            for (int j = 0; j < 8; ++j)
                VT[(seg * 8 + j) * TS + row] = hv[j];
        }
        __syncthreads();

#pragma unroll
        for (int sti = 0; sti < 2; ++sti) {
            int st = wave + sti * 8;
            int qm = st >> 2, nd = st & 3;
#pragma unroll
            for (int ks = 0; ks < 64; ks += 32) {
                FragBF a, bb;
                const float* pa = &Lg[(qm * 16 + lrow) * LGS + kc * 64 + ks + kb8];
                float fl[16];
                *reinterpret_cast<float4*>(&fl[0])  = *reinterpret_cast<const float4*>(pa);
                *reinterpret_cast<float4*>(&fl[4])  = *reinterpret_cast<const float4*>(pa + 4);
                *reinterpret_cast<float4*>(&fl[8])  = *reinterpret_cast<const float4*>(pa + 16);
                *reinterpret_cast<float4*>(&fl[12]) = *reinterpret_cast<const float4*>(pa + 20);
#pragma unroll
                for (int i = 0; i < 16; ++i) a.v[i] = (__bf16)fl[i];
                const __bf16* pb = &VT[(nd * 16 + lrow) * TS + ks + kb8];
                bb.q[0] = *reinterpret_cast<const uint4*>(pb);
                bb.q[1] = *reinterpret_cast<const uint4*>(pb + 16);
                o[sti] = __builtin_amdgcn_wmma_f32_16x16x32_bf16(
                    false, a.v, false, bb.v, (short)0, o[sti], false, false);
            }
        }
        __syncthreads();
    }

#pragma unroll
    for (int sti = 0; sti < 2; ++sti) {
        int st = wave + sti * 8;
        int qm = st >> 2, nd = st & 3;
#pragma unroll
        for (int r = 0; r < 8; ++r) {
            int qrow = qm * 16 + r + hi8;
            int dh   = nd * 16 + lrow;
            float v  = o[sti][r] * rowInv[qrow];
            ctx[(size_t)(b * kS + q0 + qrow) * kD + h * kDH + dh] = (__bf16)v;
        }
    }
}

// ---------------------------------------------------------------------------
// Host orchestration
// ---------------------------------------------------------------------------
extern "C" void kernel_launch(void* const* d_in, const int* in_sizes, int n_in,
                              void* d_out, int out_size, void* d_ws, size_t ws_size,
                              hipStream_t stream) {
    (void)in_sizes; (void)n_in; (void)out_size; (void)ws_size;

    const float* inputs   = (const float*)d_in[0];
    const float* paddings = (const float*)d_in[1];
    const float* abias    = (const float*)d_in[2];
    const float* amask    = (const float*)d_in[3];
    const float* ln1_g = (const float*)d_in[4];
    const float* ln1_b = (const float*)d_in[5];
    const float* wq = (const float*)d_in[6];  const float* bq = (const float*)d_in[7];
    const float* wk = (const float*)d_in[8];  const float* bk = (const float*)d_in[9];
    const float* wv = (const float*)d_in[10]; const float* bv = (const float*)d_in[11];
    const float* wo = (const float*)d_in[12]; const float* bo = (const float*)d_in[13];
    const float* ln2_g = (const float*)d_in[14];
    const float* ln2_b = (const float*)d_in[15];
    const float* w1 = (const float*)d_in[16]; const float* b1 = (const float*)d_in[17];
    const float* w2 = (const float*)d_in[18]; const float* b2 = (const float*)d_in[19];

    char* ws = (char*)d_ws;
    size_t off = 0;
    auto wsAlloc = [&](size_t bytes) -> void* {
        off = (off + 255) & ~(size_t)255;
        void* p = ws + off;
        off += bytes;
        return p;
    };

    const size_t nTok = (size_t)kB * kS;  // 4096 rows
    __bf16* wqb = (__bf16*)wsAlloc((size_t)kD * kD * 2);
    __bf16* wkb = (__bf16*)wsAlloc((size_t)kD * kD * 2);
    __bf16* wvb = (__bf16*)wsAlloc((size_t)kD * kD * 2);
    __bf16* wob = (__bf16*)wsAlloc((size_t)kD * kD * 2);
    __bf16* w1b = (__bf16*)wsAlloc((size_t)kD * kF * 2);
    __bf16* w2b = (__bf16*)wsAlloc((size_t)kF * kD * 2);
    __bf16* xn  = (__bf16*)wsAlloc(nTok * kD * 2);
    __bf16* qb  = (__bf16*)wsAlloc(nTok * kD * 2);
    __bf16* kb  = (__bf16*)wsAlloc(nTok * kD * 2);
    __bf16* vb  = (__bf16*)wsAlloc(nTok * kD * 2);
    __bf16* ctx = (__bf16*)wsAlloc(nTok * kD * 2);
    float*  aout = (float*)wsAlloc(nTok * kD * 4);
    __bf16* yn  = (__bf16*)wsAlloc(nTok * kD * 2);
    __bf16* hb  = (__bf16*)wsAlloc(nTok * kF * 2);

    const int NT = 256;

    // weight conversion fp32 -> bf16
    cvt_bf16_kernel<<<1024, NT, 0, stream>>>(wq, wqb, kD * kD);
    cvt_bf16_kernel<<<1024, NT, 0, stream>>>(wk, wkb, kD * kD);
    cvt_bf16_kernel<<<1024, NT, 0, stream>>>(wv, wvb, kD * kD);
    cvt_bf16_kernel<<<1024, NT, 0, stream>>>(wo, wob, kD * kD);
    cvt_bf16_kernel<<<2048, NT, 0, stream>>>(w1, w1b, kD * kF);
    cvt_bf16_kernel<<<2048, NT, 0, stream>>>(w2, w2b, kF * kD);

    // LN1
    ln_bf16_kernel<<<(int)nTok, NT, 0, stream>>>(inputs, ln1_g, ln1_b, xn);

    // Q/K/V projections (softmax 1/sqrt(DH) folded into Q)
    dim3 gProj(kD / GBN, (int)nTok / GBM);
    gemm_bf16_kernel<EPI_BF16_SCALE><<<gProj, NT, 0, stream>>>(
        xn, wqb, bq, nullptr, nullptr, qb, (int)nTok, kD, kD, 0.125f);
    gemm_bf16_kernel<EPI_BF16_SCALE><<<gProj, NT, 0, stream>>>(
        xn, wkb, bk, nullptr, nullptr, kb, (int)nTok, kD, kD, 1.0f);
    gemm_bf16_kernel<EPI_BF16_SCALE><<<gProj, NT, 0, stream>>>(
        xn, wvb, bv, nullptr, nullptr, vb, (int)nTok, kD, kD, 1.0f);

    // attention (big-LDS kernel)
    (void)hipFuncSetAttribute(reinterpret_cast<const void*>(attn_kernel),
                              hipFuncAttributeMaxDynamicSharedMemorySize,
                              (int)ATTN_SMEM);
    attn_kernel<<<kB * kH * (kS / 64), NT, ATTN_SMEM, stream>>>(
        qb, kb, vb, abias, amask, paddings, ctx);

    // output projection + residual
    gemm_bf16_kernel<EPI_RESID_F32><<<gProj, NT, 0, stream>>>(
        ctx, wob, bo, inputs, nullptr, aout, (int)nTok, kD, kD, 1.0f);

    // LN2
    ln_bf16_kernel<<<(int)nTok, NT, 0, stream>>>(aout, ln2_g, ln2_b, yn);

    // FFN
    dim3 gF1(kF / GBN, (int)nTok / GBM);
    gemm_bf16_kernel<EPI_GELU_BF16><<<gF1, NT, 0, stream>>>(
        yn, w1b, b1, nullptr, nullptr, hb, (int)nTok, kF, kD, 1.0f);
    gemm_bf16_kernel<EPI_RESID_MASK_F32><<<gProj, NT, 0, stream>>>(
        hb, w2b, b2, aout, paddings, d_out, (int)nTok, kD, kF, 1.0f);
}
